// Mixer_multi_style_85779086835816
// MI455X (gfx1250) — matrix operand order repaired and checked by reading, not compiled, not run on hardware
//
#include <hip/hip_runtime.h>

// out[a,b,k] = sum_ij s[a,b,i] c[b,j] w[i,j,k] + bias[k]
// A=8, B=I=J=K=256.
// Pass 1 (dominant, 4.3e9 MACs): V[b,i,k] = sum_j c[b,j] w[i,j,k]
//   -> 256 GEMMs (one per i): V_i = C (256x256) x W_i (256x256), bf16 WMMA, f32 accum.
//   V stored TRANSPOSED as bf16 Vt[b][k][i] so pass 2 B-fragments are contiguous.
// Pass 2 (1.3e8 MACs): out[a,b,k] = sum_i s[a,b,i] V[b,i,k] + bias[k], bf16 WMMA.

typedef __attribute__((ext_vector_type(16))) __bf16       v16bf;
typedef __attribute__((ext_vector_type(8)))  float        v8f;
typedef __attribute__((ext_vector_type(4)))  unsigned int v4u;

union V16 { v16bf bf; v4u u4[2]; unsigned short us[16]; };
union F4  { float4 v; float a[4]; };

__device__ __forceinline__ unsigned short f2bf(float x) {
  unsigned int u = __float_as_uint(x);
  u += 0x7FFFu + ((u >> 16) & 1u);        // round-to-nearest-even
  return (unsigned short)(u >> 16);
}

#define LP 40   // LDS row pitch in bf16 elems (80 B): 16B-aligned b128 frags, bank-spread

// ---------------- Pass 1: V_i = C x W_i --------------------------------------
__global__ __launch_bounds__(256) void pass1_cw(
    const float* __restrict__ Cmat,        // input_c [256 b][256 j]
    const float* __restrict__ W,           // w [256 i][256 j][256 k]
    unsigned short* __restrict__ Vt,       // bf16 [256 b][kspan][256 i]
    int kbase, int kspan)
{
  const int tid  = threadIdx.x;
  const int lane = tid & 31;
  const int wave = tid >> 5;
  const int bt   = blockIdx.x & 1;         // b half
  const int kt   = blockIdx.x >> 1;        // k 128-tile within chunk
  const int i    = blockIdx.y;             // 0..255
  const int b0   = bt * 128;
  const int k0g  = kbase + kt * 128;       // global k base of this tile
  const int k0l  = kt * 128;               // k base within chunk

  __shared__ unsigned short lC[128 * LP];  // A: [b_local][j_local]
  __shared__ unsigned short lW[128 * LP];  // B: transposed [k_local][j_local]

  v8f acc[8] = {};
  const float* Wi = W + (size_t)i * (256 * 256);

  for (int js = 0; js < 256; js += 32) {
    // ---- stage C tile [128 b x 32 j] -> bf16 LDS (natural order)
    #pragma unroll
    for (int it = 0; it < 4; ++it) {
      const int r = (tid >> 3) + it * 32;
      const int c = (tid & 7) * 4;
      F4 f; f.v = *(const float4*)(Cmat + (size_t)(b0 + r) * 256 + js + c);
      *(ushort4*)(&lC[r * LP + c]) =
          make_ushort4(f2bf(f.a[0]), f2bf(f.a[1]), f2bf(f.a[2]), f2bf(f.a[3]));
    }
    // ---- stage W tile [32 j x 128 k] -> LDS transposed Wt[k][j] via 4x4 blocks
    {
      const int kb = (tid & 31) * 4;       // coalesced along k
      const int jb = (tid >> 5) * 4;
      F4 f[4];
      #pragma unroll
      for (int c = 0; c < 4; ++c)
        f[c].v = *(const float4*)(Wi + (size_t)(js + jb + c) * 256 + k0g + kb);
      #pragma unroll
      for (int rr = 0; rr < 4; ++rr)
        *(ushort4*)(&lW[(kb + rr) * LP + jb]) =
            make_ushort4(f2bf(f[0].a[rr]), f2bf(f[1].a[rr]),
                         f2bf(f[2].a[rr]), f2bf(f[3].a[rr]));
      if (js + 32 < 256)   // global_prefetch of next j-slab of w
        __builtin_prefetch(Wi + (size_t)(js + 32 + jb) * 256 + k0g + kb, 0, 0);
    }
    __syncthreads();

    // ---- compute: each wave owns a 16-row strip x 128 cols (8 WMMA tiles)
    // A frag (16-bit A 16x32 layout): lane = row, K groups (L>>4)*8 and +16
    const int arow = wave * 16 + (lane & 15);
    const int ja   = (lane >> 4) * 8;
    V16 a;
    a.u4[0] = *(const v4u*)(&lC[arow * LP + ja]);
    a.u4[1] = *(const v4u*)(&lC[arow * LP + 16 + ja]);
    #pragma unroll
    for (int n = 0; n < 8; ++n) {
      // B frag (32x16): lane = column, K groups (L>>4)*8 and +16, pairs packed per VGPR
      const int col = n * 16 + (lane & 15);
      const int jb2 = (lane >> 4) * 8;
      V16 bb;
      bb.u4[0] = *(const v4u*)(&lW[col * LP + jb2]);
      bb.u4[1] = *(const v4u*)(&lW[col * LP + jb2 + 16]);
      acc[n] = __builtin_amdgcn_wmma_f32_16x16x32_bf16(
          false, a.bf, false, bb.bf, (short)0, acc[n], false, false);
    }
    __syncthreads();
  }

  // ---- epilogue: D rows: VGPR r -> M=r (lanes 0-15) / M=8+r (lanes 16-31)
  const int mofs = (lane >> 4) * 8;
  const int colb = lane & 15;
  #pragma unroll
  for (int n = 0; n < 8; ++n) {
    const int kl = k0l + n * 16 + colb;
    #pragma unroll
    for (int r = 0; r < 8; ++r) {
      const int brow = b0 + wave * 16 + mofs + r;
      Vt[((size_t)brow * kspan + kl) * 256 + i] = f2bf(acc[n][r]);
    }
  }
}

// ---------------- Pass 2: out_b = S_b x V_b + bias ---------------------------
__global__ __launch_bounds__(256) void pass2_sv(
    const float* __restrict__ S,            // input_s [8 a][256 b][256 i]
    const unsigned short* __restrict__ Vt,  // bf16 [256 b][kspan][256 i]
    const float* __restrict__ bias,         // [256]
    float* __restrict__ out,                // [8 a][256 b][256 k]
    int kbase, int kspan)
{
  const int b    = blockIdx.x;
  const int tid  = threadIdx.x;
  const int lane = tid & 31;
  const int wave = tid >> 5;
  const int ntpw = kspan >> 7;              // k-tiles per wave: 1 or 2 (uniform)

  v8f acc[2] = {};
  // A rows 8..15 duplicate rows 0..7 (their D rows are never stored)
  const int arow = lane & 7;
  const int ja   = (lane >> 4) * 8;
  const float* sp = S + ((size_t)arow * 256 + b) * 256;

  for (int is = 0; is < 256; is += 32) {
    F4 f0, f1, f2, f3;
    f0.v = *(const float4*)(sp + is + ja);
    f1.v = *(const float4*)(sp + is + ja + 4);
    f2.v = *(const float4*)(sp + is + 16 + ja);
    f3.v = *(const float4*)(sp + is + 16 + ja + 4);
    V16 a;
    #pragma unroll
    for (int q = 0; q < 4; ++q) {
      a.us[q]      = f2bf(f0.a[q]);
      a.us[4 + q]  = f2bf(f1.a[q]);
      a.us[8 + q]  = f2bf(f2.a[q]);
      a.us[12 + q] = f2bf(f3.a[q]);
    }
    #pragma unroll
    for (int t = 0; t < 2; ++t) {
      if (t < ntpw) {                       // wave-uniform: EXEC stays all-1s
        const int kl = (wave * ntpw + t) * 16 + (lane & 15);
        const unsigned short* vp =
            Vt + ((size_t)b * kspan + kl) * 256 + is + (lane >> 4) * 8;
        V16 bb;
        bb.u4[0] = *(const v4u*)(vp);
        bb.u4[1] = *(const v4u*)(vp + 16);
        acc[t] = __builtin_amdgcn_wmma_f32_16x16x32_bf16(
            false, a.bf, false, bb.bf, (short)0, acc[t], false, false);
      }
    }
  }

  const int m0 = (lane >> 4) * 8;
  #pragma unroll
  for (int t = 0; t < 2; ++t) {
    if (t < ntpw) {
      const int kl = (wave * ntpw + t) * 16 + (lane & 15);
      const int kg = kbase + kl;
      const float bv = bias[kg];
      #pragma unroll
      for (int r = 0; r < 8; ++r) {
        const int aa = m0 + r;
        if (aa < 8)
          out[((size_t)aa * 256 + b) * 256 + kg] = acc[t][r] + bv;
      }
    }
  }
}

extern "C" void kernel_launch(void* const* d_in, const int* in_sizes, int n_in,
                              void* d_out, int out_size, void* d_ws, size_t ws_size,
                              hipStream_t stream) {
  (void)in_sizes; (void)n_in; (void)out_size;
  const float* S    = (const float*)d_in[0];   // input_s (8,256,256)
  const float* C    = (const float*)d_in[1];   // input_c (256,256)
  const float* W    = (const float*)d_in[2];   // w (256,256,256)
  const float* bias = (const float*)d_in[3];   // b (256,)
  float* out = (float*)d_out;
  unsigned short* Vt = (unsigned short*)d_ws;

  const size_t need_full = (size_t)256 * 256 * 256 * sizeof(unsigned short); // 32 MiB
  const int nchunk = (ws_size >= need_full) ? 1 : 2;   // fall back to 16 MiB ws
  const int kspan  = 256 / nchunk;
  for (int c = 0; c < nchunk; ++c) {
    const int kb = c * kspan;
    dim3 g1(2 * (kspan / 128), 256);
    pass1_cw<<<g1, 256, 0, stream>>>(C, W, Vt, kb, kspan);
    pass2_sv<<<dim3(256), 256, 0, stream>>>(S, Vt, bias, out, kb, kspan);
  }
}
